// SparseOPTAttention_71253507441025
// MI455X (gfx1250) — compile-verified, matrix operand order checked by reading, not run on hardware
//
#include <hip/hip_runtime.h>
#include <hip/hip_bf16.h>
#include <stdint.h>

typedef __attribute__((ext_vector_type(32))) __bf16 v32bf;
typedef __attribute__((ext_vector_type(16))) __bf16 v16bf;
typedef __attribute__((ext_vector_type(8)))  __bf16 v8bf;
typedef __attribute__((ext_vector_type(8)))  float  v8f;

#define DME   2048
#define BSZ   2
#define TSEQ  2048
#define NH    32
#define HD    64
#define BT    (BSZ * TSEQ)
#define KC    (DME / 2)      /* compressed K length = 1024 */
#define NKG   (DME / 32)     /* index words per row = 64   */
#define LOG2E 1.4426950408889634f

// ---------------------------------------------------------------- helpers

__device__ __forceinline__ __bf16 to_bf16(float f) {
  union { float f; uint32_t u; } c; c.f = f;
  uint32_t r = c.u + 0x7FFFu + ((c.u >> 16) & 1u);   // round-nearest-even
  uint16_t h = (uint16_t)(r >> 16);
  __bf16 o; __builtin_memcpy(&o, &h, 2);
  return o;
}

// gfx1250 async global -> LDS copy of 16 bytes (tracked by ASYNCcnt)
__device__ __forceinline__ void async_cp16(uint32_t lds_off, const void* g) {
  asm volatile("global_load_async_to_lds_b128 %0, %1, off"
               :: "v"(lds_off), "v"((uint64_t)(uintptr_t)g) : "memory");
}
__device__ __forceinline__ void async_wait0() {
  asm volatile("s_wait_asynccnt 0" ::: "memory");
}

// 16x32 bf16 fragment (dense A / 32x16 B) from a row-major tile.
__device__ __forceinline__ v16bf ld_frag(const __bf16* base, int ld, int kbase) {
  int lane = threadIdx.x & 31;
  int r = lane & 15, half = lane >> 4;
  const __bf16* p = base + (size_t)r * ld + kbase + half * 8;
  v8bf a = *(const v8bf*)p;
  v8bf b = *(const v8bf*)(p + 16);
  return __builtin_shufflevector(a, b, 0,1,2,3,4,5,6,7,8,9,10,11,12,13,14,15);
}

// 64x16 bf16 B fragment (for K=64 SWMMAC): per ISA, lane<16 -> col=lane,
// K runs {0..15, 32..47}; lane>=16 -> col=lane-16, K runs {16..31, 48..63}.
__device__ __forceinline__ v32bf ld_bfrag64(const __bf16* base, int ld) {
  int lane = threadIdx.x & 31;
  int col = lane & 15, half = lane >> 4;
  const __bf16* p = base + (size_t)col * ld + half * 16;
  v8bf r0 = *(const v8bf*)p;
  v8bf r1 = *(const v8bf*)(p + 8);
  v8bf r2 = *(const v8bf*)(p + 32);
  v8bf r3 = *(const v8bf*)(p + 40);
  v16bf lo = __builtin_shufflevector(r0, r1, 0,1,2,3,4,5,6,7,8,9,10,11,12,13,14,15);
  v16bf hi = __builtin_shufflevector(r2, r3, 0,1,2,3,4,5,6,7,8,9,10,11,12,13,14,15);
  return __builtin_shufflevector(lo, hi,
      0,1,2,3,4,5,6,7,8,9,10,11,12,13,14,15,
      16,17,18,19,20,21,22,23,24,25,26,27,28,29,30,31);
}

__device__ __forceinline__ v8f wmma_bf16(v16bf a, v16bf b, v8f c) {
  return __builtin_amdgcn_wmma_f32_16x16x32_bf16(false, a, false, b,
                                                 (short)0, c, false, false);
}

// 2:4 sparse MAC: D(=C) += expand(A, idx) x B   (A 16x32 compressed, K=64)
__device__ __forceinline__ void swmmac_bf16(v8f& d, v16bf a, v32bf b, uint32_t idx) {
  asm volatile("v_swmmac_f32_16x16x64_bf16 %0, %1, %2, %3"
               : "+v"(d) : "v"(a), "v"(b), "v"(idx));
}

// ---------------------------------------------------------------- prep kernels

// perm is one-hot DxD: p[j] = original column at permuted position j,
// pinv[p[j]] = j.
__global__ void k_perm_extract(const float* __restrict__ perm,
                               int* __restrict__ p, int* __restrict__ pinv) {
  int j = blockIdx.x * blockDim.x + threadIdx.x;
  if (j >= DME) return;
  int best = 0; float bv = -1.f;
  for (int i = 0; i < DME; ++i) {
    float v = perm[(size_t)i * DME + j];
    if (v > bv) { bv = v; best = i; }
  }
  p[j] = best;
  pinv[best] = j;
}

// xp[m][j] = bf16(x[m][p(j)])  — activations in permuted-K order
__global__ void k_permcast(const float* __restrict__ x, const int* __restrict__ p,
                           __bf16* __restrict__ xp) {
  int i = blockIdx.x * blockDim.x + threadIdx.x;
  if (i >= BT * DME) return;
  int m = i >> 11, j = i & (DME - 1);
  xp[i] = to_bf16(x[(size_t)m * DME + p[j]]);
}

// Wanda top-2-of-4 in permuted-K order; emit compressed bf16 values +
// ISA-layout 2-bit index words (4 bits per 4-wide block, idx0 < idx1).
__global__ void k_pack(const float* __restrict__ W, const float* __restrict__ scaler,
                       const int* __restrict__ p, __bf16* __restrict__ Wc,
                       uint32_t* __restrict__ Widx) {
  int idx = blockIdx.x * blockDim.x + threadIdx.x;   // row * NKG + kgroup
  if (idx >= DME * NKG) return;
  int r = idx >> 6, kg = idx & (NKG - 1);
  uint32_t word = 0;
#pragma unroll
  for (int bl = 0; bl < 8; ++bl) {
    int jb = kg * 32 + bl * 4;
    float w[4], met[4];
#pragma unroll
    for (int t = 0; t < 4; ++t) {
      int c = p[jb + t];
      w[t] = W[(size_t)r * DME + c];
      met[t] = fabsf(w[t]) * sqrtf(scaler[c]);
    }
    int i0 = 0;
#pragma unroll
    for (int t = 1; t < 4; ++t) if (met[t] > met[i0]) i0 = t;
    int i1 = -1;
#pragma unroll
    for (int t = 0; t < 4; ++t)
      if (t != i0 && (i1 < 0 || met[t] > met[i1])) i1 = t;
    if (i0 > i1) { int tmp = i0; i0 = i1; i1 = tmp; }
    Wc[(size_t)r * KC + kg * 16 + bl * 2 + 0] = to_bf16(w[i0]);
    Wc[(size_t)r * KC + kg * 16 + bl * 2 + 1] = to_bf16(w[i1]);
    word |= (uint32_t)(i0 | (i1 << 2)) << (bl * 4);
  }
  Widx[(size_t)r * NKG + kg] = word;
}

// ---------------------------------------------------------------- sparse GEMM

// Y^T tile: D(16 feat x 16 rows) = A(Wc sparse, K=64 expanded) x B(Xp^T).
// Block tile: 128 features x 128 rows, K-step 64, double-buffered async LDS.
// LDS: A 2x(128x32 bf16)=16K at 0 | B 2x(128x64 bf16)=32K at 16384. 48KB.
__device__ __forceinline__ void stage_sp(const __bf16* Wc, const __bf16* Xp,
                                         int n0, int m0, int ks, int buf, int tid) {
  uint32_t abase = (uint32_t)buf * 8192u;
  uint32_t bbase = 16384u + (uint32_t)buf * 16384u;
#pragma unroll
  for (int c = 0; c < 2; ++c) {
    int ch = tid + c * 256;
    int row = ch >> 2, kc = (ch & 3) * 8;
    async_cp16(abase + (uint32_t)(row * 32 + kc) * 2u,
               Wc + (size_t)(n0 + row) * KC + ks * 32 + kc);
  }
#pragma unroll
  for (int c = 0; c < 4; ++c) {
    int ch = tid + c * 256;
    int row = ch >> 3, kc = (ch & 7) * 8;
    async_cp16(bbase + (uint32_t)(row * 64 + kc) * 2u,
               Xp + (size_t)(m0 + row) * DME + ks * 64 + kc);
  }
}

__global__ __launch_bounds__(256)
void k_gemm_sp(const __bf16* __restrict__ Wc, const uint32_t* __restrict__ Widx,
               const __bf16* __restrict__ Xp, const float* __restrict__ bias,
               float scale, int headmode,
               __bf16* __restrict__ obf, float* __restrict__ of) {
  extern __shared__ char smem[];
  int tid = threadIdx.x, lane = tid & 31, wid = tid >> 5;
  int n0 = blockIdx.x * 128, m0 = blockIdx.y * 128;
  int wn = wid & 1, wm = wid >> 1;            // 2 (feat) x 4 (rows) wave grid
  int col = lane & 15, half = lane >> 4;

  v8f acc[4][2];
#pragma unroll
  for (int i = 0; i < 4; ++i)
#pragma unroll
    for (int j = 0; j < 2; ++j)
#pragma unroll
      for (int g = 0; g < 8; ++g) acc[i][j][g] = 0.f;

  __builtin_prefetch(Widx + (size_t)(n0 + wn * 64 + col) * NKG, 0, 0);

  stage_sp(Wc, Xp, n0, m0, 0, 0, tid);
  async_wait0();
  __syncthreads();

  const int KSTEPS = DME / 64;
  for (int ks = 0; ks < KSTEPS; ++ks) {
    int buf = ks & 1;
    if (ks + 1 < KSTEPS)
      stage_sp(Wc, Xp, n0, m0, ks + 1, buf ^ 1, tid);

    const __bf16* Ab = (const __bf16*)smem + buf * 4096;
    const __bf16* Bb = (const __bf16*)(smem + 16384) + buf * 8192;

    v16bf af[4]; uint32_t ai[4]; v32bf bfv[2];
#pragma unroll
    for (int tm = 0; tm < 4; ++tm) {
      af[tm] = ld_frag(Ab + (wn * 64 + tm * 16) * 32, 32, 0);
      ai[tm] = Widx[(size_t)(n0 + wn * 64 + tm * 16 + col) * NKG + ks * 2 + half];
    }
#pragma unroll
    for (int tr = 0; tr < 2; ++tr)
      bfv[tr] = ld_bfrag64(Bb + (wm * 32 + tr * 16) * 64, 64);
#pragma unroll
    for (int tm = 0; tm < 4; ++tm)
#pragma unroll
      for (int tr = 0; tr < 2; ++tr)
        swmmac_bf16(acc[tm][tr], af[tm], bfv[tr], ai[tm]);

    async_wait0();
    __syncthreads();
  }

  // epilogue: D rows are output features n, D cols are batch rows m
#pragma unroll
  for (int tm = 0; tm < 4; ++tm)
#pragma unroll
    for (int tr = 0; tr < 2; ++tr) {
      int m = m0 + wm * 32 + tr * 16 + col;
#pragma unroll
      for (int g = 0; g < 8; ++g) {
        int n = n0 + wn * 64 + tm * 16 + half * 8 + g;
        float r = (acc[tm][tr][g] + bias[n]) * scale;
        if (headmode) {                       // [B,H,T,hd] bf16
          int bb = m >> 11, t = m & (TSEQ - 1);
          int h = n >> 6,  d = n & (HD - 1);
          obf[(((size_t)bb * NH + h) * TSEQ + t) * HD + d] = to_bf16(r);
        } else {                              // [BT, D] f32
          of[(size_t)m * DME + n] = r;
        }
      }
    }
}

// ---------------------------------------------------------------- flash attention

// One block = (b, h, 128 q-rows); each wave owns 16 q-rows.
// LDS: Kt[64][64] (8K) | Vt[64][64] transposed (8K) | Pw 8 x 16x64 (16K)
__global__ __launch_bounds__(256)
void k_attn(const __bf16* __restrict__ Q, const __bf16* __restrict__ Kb,
            const __bf16* __restrict__ Vb, const float* __restrict__ amask,
            const int* __restrict__ invo, __bf16* __restrict__ ctxp) {
  extern __shared__ char smem[];
  __bf16* Kt = (__bf16*)smem;
  __bf16* Vt = (__bf16*)(smem + 8192);
  int tid = threadIdx.x, lane = tid & 31, wid = tid >> 5;
  __bf16* Pw = (__bf16*)(smem + 16384) + wid * (16 * 64);

  int b = blockIdx.z, h = blockIdx.y;
  int q0 = blockIdx.x * 128 + wid * 16;
  size_t headoff = ((size_t)b * NH + h) * TSEQ;
  const __bf16* qptr = Q + (headoff + q0) * HD;
  v16bf qa0 = ld_frag(qptr, HD, 0);
  v16bf qa1 = ld_frag(qptr, HD, 32);
  int half = lane >> 4, col = lane & 15;

  float mrun[8], lrun[8];
  v8f oacc[4];
#pragma unroll
  for (int g = 0; g < 8; ++g) { mrun[g] = -3.0e38f; lrun[g] = 0.f; }
#pragma unroll
  for (int dt = 0; dt < 4; ++dt)
#pragma unroll
    for (int g = 0; g < 8; ++g) oacc[dt][g] = 0.f;

  for (int kb = 0; kb < TSEQ / 64; ++kb) {
#pragma unroll
    for (int c = 0; c < 2; ++c) {
      int ch = tid + c * 256;
      int r = ch >> 3, kc = (ch & 7) * 8;
      async_cp16((uint32_t)(r * 64 + kc) * 2u,
                 Kb + (headoff + kb * 64 + r) * HD + kc);
    }
    {
      int vr = tid >> 2, cs = (tid & 3) * 16;
      const __bf16* vp = Vb + (headoff + kb * 64 + vr) * HD + cs;
      v8bf v0 = *(const v8bf*)vp;
      v8bf v1 = *(const v8bf*)(vp + 8);
#pragma unroll
      for (int i = 0; i < 8; ++i) {
        Vt[(cs + i) * 64 + vr]     = v0[i];
        Vt[(cs + 8 + i) * 64 + vr] = v1[i];
      }
    }
    async_wait0();
    __syncthreads();

    v8f sacc[4];
#pragma unroll
    for (int t = 0; t < 4; ++t) {
#pragma unroll
      for (int g = 0; g < 8; ++g) sacc[t][g] = 0.f;
      v16bf kf0 = ld_frag(Kt + (t * 16) * 64, 64, 0);
      sacc[t] = wmma_bf16(qa0, kf0, sacc[t]);
      v16bf kf1 = ld_frag(Kt + (t * 16) * 64, 64, 32);
      sacc[t] = wmma_bf16(qa1, kf1, sacc[t]);
    }
#pragma unroll
    for (int t = 0; t < 4; ++t)
#pragma unroll
      for (int g = 0; g < 8; ++g) {
        int qr = q0 + half * 8 + g;
        int kc = kb * 64 + t * 16 + col;
        float s = sacc[t][g] + amask[((size_t)b * TSEQ + qr) * TSEQ + kc];
        sacc[t][g] = fmaxf(s, -3.4028235e38f);
      }
    float fac[8];
#pragma unroll
    for (int g = 0; g < 8; ++g) {
      float mx = fmaxf(fmaxf(sacc[0][g], sacc[1][g]), fmaxf(sacc[2][g], sacc[3][g]));
#pragma unroll
      for (int m = 8; m >= 1; m >>= 1) mx = fmaxf(mx, __shfl_xor(mx, m, 32));
      float mn = fmaxf(mrun[g], mx);
      fac[g] = exp2f((mrun[g] - mn) * LOG2E);
      mrun[g] = mn;
    }
    float rs[8];
#pragma unroll
    for (int g = 0; g < 8; ++g) rs[g] = 0.f;
#pragma unroll
    for (int t = 0; t < 4; ++t)
#pragma unroll
      for (int g = 0; g < 8; ++g) {
        float p = exp2f((sacc[t][g] - mrun[g]) * LOG2E);
        sacc[t][g] = p; rs[g] += p;
      }
#pragma unroll
    for (int g = 0; g < 8; ++g) {
      float r = rs[g];
#pragma unroll
      for (int m = 8; m >= 1; m >>= 1) r += __shfl_xor(r, m, 32);
      lrun[g] = lrun[g] * fac[g] + r;
    }
#pragma unroll
    for (int dt = 0; dt < 4; ++dt)
#pragma unroll
      for (int g = 0; g < 8; ++g) oacc[dt][g] *= fac[g];

#pragma unroll
    for (int t = 0; t < 4; ++t)
#pragma unroll
      for (int g = 0; g < 8; ++g)
        Pw[(half * 8 + g) * 64 + t * 16 + col] = to_bf16(sacc[t][g]);

#pragma unroll
    for (int kstep = 0; kstep < 2; ++kstep) {
      v16bf pa = ld_frag(Pw, 64, kstep * 32);
#pragma unroll
      for (int dt = 0; dt < 4; ++dt) {
        v16bf vf = ld_frag(Vt + (dt * 16) * 64, 64, kstep * 32);
        oacc[dt] = wmma_bf16(pa, vf, oacc[dt]);
      }
    }
    __syncthreads();
  }

  // write context in perm_o-permuted column order (ready for sparse o-proj)
#pragma unroll
  for (int dt = 0; dt < 4; ++dt)
#pragma unroll
    for (int g = 0; g < 8; ++g) {
      int qr = q0 + half * 8 + g;
      int d  = h * HD + dt * 16 + col;
      float o = oacc[dt][g] / lrun[g];
      ctxp[((size_t)b * TSEQ + qr) * DME + invo[d]] = to_bf16(o);
    }
}

// ---------------------------------------------------------------- launch

extern "C" void kernel_launch(void* const* d_in, const int* in_sizes, int n_in,
                              void* d_out, int out_size, void* d_ws, size_t ws_size,
                              hipStream_t stream) {
  const float* x     = (const float*)d_in[0];
  const float* amask = (const float*)d_in[1];
  const float* Wq = (const float*)d_in[2];  const float* bq = (const float*)d_in[3];
  const float* Wk = (const float*)d_in[4];  const float* bk = (const float*)d_in[5];
  const float* Wv = (const float*)d_in[6];  const float* bv = (const float*)d_in[7];
  const float* Wo = (const float*)d_in[8];  const float* bo = (const float*)d_in[9];
  const float* sq = (const float*)d_in[10]; const float* sk = (const float*)d_in[11];
  const float* sv = (const float*)d_in[12]; const float* so = (const float*)d_in[13];
  const float* permqkv = (const float*)d_in[14];
  const float* permo   = (const float*)d_in[15];
  float* out = (float*)d_out;

  char* w = (char*)d_ws;
  int* p_qkv   = (int*)w;          w += 8192;
  int* pinv_qkv= (int*)w;          w += 8192;
  int* p_o     = (int*)w;          w += 8192;
  int* pinv_o  = (int*)w;          w += 8192;
  const size_t WCBYTES = (size_t)DME * KC * 2;       // 4 MB compressed weights
  const size_t WIBYTES = (size_t)DME * NKG * 4;      // 512 KB index words
  __bf16*   Wqc = (__bf16*)w;      w += WCBYTES;
  __bf16*   Wkc = (__bf16*)w;      w += WCBYTES;
  __bf16*   Wvc = (__bf16*)w;      w += WCBYTES;
  __bf16*   Woc = (__bf16*)w;      w += WCBYTES;
  uint32_t* Wqi = (uint32_t*)w;    w += WIBYTES;
  uint32_t* Wki = (uint32_t*)w;    w += WIBYTES;
  uint32_t* Wvi = (uint32_t*)w;    w += WIBYTES;
  uint32_t* Woi = (uint32_t*)w;    w += WIBYTES;
  const size_t ABYTES = (size_t)BT * DME * 2;        // 16 MB activations
  __bf16* xp   = (__bf16*)w;       w += ABYTES;
  __bf16* qb   = (__bf16*)w;       w += ABYTES;
  __bf16* kbuf = (__bf16*)w;       w += ABYTES;
  __bf16* vbuf = (__bf16*)w;       w += ABYTES;
  __bf16* ctxp = (__bf16*)w;       w += ABYTES;

  // 1) permutation index + inverse extraction
  k_perm_extract<<<(DME + 255) / 256, 256, 0, stream>>>(permqkv, p_qkv, pinv_qkv);
  k_perm_extract<<<(DME + 255) / 256, 256, 0, stream>>>(permo,   p_o,   pinv_o);

  // 2) x -> bf16 in permuted-K order
  int nel = BT * DME;
  k_permcast<<<(nel + 255) / 256, 256, 0, stream>>>(x, p_qkv, xp);

  // 3) Wanda 2:4 pack (compressed values + sparse index words)
  int npk = DME * NKG;
  k_pack<<<(npk + 255) / 256, 256, 0, stream>>>(Wq, sq, p_qkv, Wqc, Wqi);
  k_pack<<<(npk + 255) / 256, 256, 0, stream>>>(Wk, sk, p_qkv, Wkc, Wki);
  k_pack<<<(npk + 255) / 256, 256, 0, stream>>>(Wv, sv, p_qkv, Wvc, Wvi);
  k_pack<<<(npk + 255) / 256, 256, 0, stream>>>(Wo, so, p_o,   Woc, Woi);

  // 4) sparse Q/K/V projections (Q carries 1/sqrt(hd) = 0.125)
  dim3 ggrid(DME / 128, BT / 128);
  k_gemm_sp<<<ggrid, 256, 49152, stream>>>(Wqc, Wqi, xp, bq, 0.125f, 1, qb,   nullptr);
  k_gemm_sp<<<ggrid, 256, 49152, stream>>>(Wkc, Wki, xp, bk, 1.0f,   1, kbuf, nullptr);
  k_gemm_sp<<<ggrid, 256, 49152, stream>>>(Wvc, Wvi, xp, bv, 1.0f,   1, vbuf, nullptr);

  // 5) flash attention (writes permuted context)
  dim3 agrid(TSEQ / 128, NH, BSZ);
  k_attn<<<agrid, 256, 32768, stream>>>(qb, kbuf, vbuf, amask, pinv_o, ctxp);

  // 6) sparse output projection -> f32 d_out
  k_gemm_sp<<<ggrid, 256, 49152, stream>>>(Woc, Woi, ctxp, bo, 1.0f, 0, nullptr, out);
}
